// EnhancedMobileViTBlock_87651692577377
// MI455X (gfx1250) — compile-verified, matrix operand order checked by reading, not run on hardware
//
#include <hip/hip_runtime.h>
#include <math.h>

// ---------------------------------------------------------------- constants
#define B_     8
#define C_     256
#define H_     32
#define W_     32
#define HW_    1024
#define NTOK   1024      // tokens per image (16*16 patches * 2*2)
#define HEADS_ 8
#define DH_    32
#define HID_   768
#define MROWS  8192      // B_*NTOK == B_*HW_

typedef _Float16 h16;
typedef __attribute__((ext_vector_type(16))) _Float16 v16h;
typedef __attribute__((ext_vector_type(8)))  float    v8f;
typedef __attribute__((ext_vector_type(4)))  int      i32x4;

#define WMMA_F16(a, b, c) \
  __builtin_amdgcn_wmma_f32_16x16x32_f16(false, (a), false, (b), (short)0, (c), false, false)

// Optional gfx1250 async global->LDS path (ASYNCcnt), guarded so either
// toolchain compiles. Builtin signature (from hipcc diagnostic):
//   (AS1 int4* src, AS3 int4* dst, int offset, int cpol)
#define AS1 __attribute__((address_space(1)))
#define AS3 __attribute__((address_space(3)))
#if defined(__has_builtin)
#if __has_builtin(__builtin_amdgcn_global_load_async_to_lds_b128) && \
    __has_builtin(__builtin_amdgcn_s_wait_asynccnt)
#define HAVE_ASYNC_LDS 1
#endif
#endif

// ------------------------------------------------- WMMA fragment loader
// Loads a 16x32 f16 tile (rows x K) in the CDNA5 A/B-matrix VGPR layout:
//  lane = hs*16 + r ; VGPR i in [0,3]: K = hs*8 + 2i, +1 ; VGPR i in [4,7]: K = 16 + hs*8 + 2(i-4), +1
// p0 points at element (row 0, k 0) of the tile; ld = row stride in halves.
__device__ inline v16h load_frag(const h16* __restrict__ p0, int ld) {
  const int lane = threadIdx.x & 31;
  const int hs = lane >> 4;
  const h16* p = p0 + (size_t)(lane & 15) * ld;
  union { v16h v; unsigned int u[8]; } f;
#pragma unroll
  for (int i = 0; i < 4; ++i)
    f.u[i] = *reinterpret_cast<const unsigned int*>(p + hs * 8 + 2 * i);
#pragma unroll
  for (int i = 0; i < 4; ++i)
    f.u[4 + i] = *reinterpret_cast<const unsigned int*>(p + 16 + hs * 8 + 2 * i);
  return f.v;
}

// ------------------------------------------------- helpers
__device__ inline float gelu_f(float v) {
  return 0.5f * v * (1.f + erff(v * 0.70710678118654752f));
}
__device__ inline float redsum32(float v) {
#pragma unroll
  for (int m = 1; m < 32; m <<= 1) v += __shfl_xor(v, m, 32);
  return v;
}
__device__ inline float redsum16(float v) {
#pragma unroll
  for (int m = 1; m < 16; m <<= 1) v += __shfl_xor(v, m, 32);
  return v;
}
__device__ inline float redmax16(float v) {
#pragma unroll
  for (int m = 1; m < 16; m <<= 1) v = fmaxf(v, __shfl_xor(v, m, 32));
  return v;
}

// ------------------------------------------------- generic WMMA GEMM
// D[M,N] = A[M,K](f16) * Wt[N,K]^T(f16) + bias, with fused epilogues.
// Block: 256 threads = 8 waves, block tile 64(M) x 128(N), wave tile 32x32.
// EPI: 0=f32 out, 1=f32 out + residual, 2=BN+GELU f32 out,
//      3=f16 out, 4=sigmoid(acc)*other -> f16 out, 5=f32 out stored NCHW.
template <int EPI>
__global__ __launch_bounds__(256) void gemm_wmma(
    const h16* __restrict__ A, const h16* __restrict__ Wt,
    const float* __restrict__ bias,
    float* __restrict__ out, h16* __restrict__ outh,
    const float* __restrict__ res, const float* __restrict__ other,
    const float* __restrict__ bng, const float* __restrict__ bnb,
    const float* __restrict__ bnm, const float* __restrict__ bnv,
    int M, int N, int K) {
  const int lane = threadIdx.x & 31;
  const int wv = threadIdx.x >> 5;
  const int m0 = blockIdx.x * 64 + (wv >> 2) * 32;
  const int n0 = blockIdx.y * 128 + (wv & 3) * 32;

  v8f acc[2][2];
#pragma unroll
  for (int i = 0; i < 2; ++i)
#pragma unroll
    for (int j = 0; j < 2; ++j) acc[i][j] = (v8f){};

  for (int k0 = 0; k0 < K; k0 += 32) {
    __builtin_prefetch(A + (size_t)m0 * K + k0 + 64, 0, 1);
    v16h a0 = load_frag(A + (size_t)m0 * K + k0, K);
    v16h a1 = load_frag(A + (size_t)(m0 + 16) * K + k0, K);
    v16h b0 = load_frag(Wt + (size_t)n0 * K + k0, K);
    v16h b1 = load_frag(Wt + (size_t)(n0 + 16) * K + k0, K);
    acc[0][0] = WMMA_F16(a0, b0, acc[0][0]);
    acc[0][1] = WMMA_F16(a0, b1, acc[0][1]);
    acc[1][0] = WMMA_F16(a1, b0, acc[1][0]);
    acc[1][1] = WMMA_F16(a1, b1, acc[1][1]);
  }

  const int hs = lane >> 4, nl = lane & 15;
#pragma unroll
  for (int mi = 0; mi < 2; ++mi)
#pragma unroll
    for (int ni = 0; ni < 2; ++ni) {
      v8f c = acc[mi][ni];
#pragma unroll
      for (int r = 0; r < 8; ++r) {
        int m = m0 + mi * 16 + r + 8 * hs;
        int n = n0 + ni * 16 + nl;
        if (m >= M || n >= N) continue;
        float val = c[r] + bias[n];
        size_t off = (size_t)m * N + n;
        if constexpr (EPI == 0) {
          out[off] = val;
        } else if constexpr (EPI == 1) {
          out[off] = val + res[off];
        } else if constexpr (EPI == 2) {
          float u = (val - bnm[n]) * rsqrtf(bnv[n] + 1e-5f) * bng[n] + bnb[n];
          out[off] = gelu_f(u);
        } else if constexpr (EPI == 3) {
          outh[off] = (h16)val;
        } else if constexpr (EPI == 4) {
          float gz = 1.f / (1.f + __expf(-val));
          outh[off] = (h16)(gz * other[off]);
        } else if constexpr (EPI == 5) {
          int bb = m >> 10, hw = m & 1023;          // m = b*HW + hw
          out[((size_t)(bb * C_ + n) << 10) + hw] = val;
        }
      }
    }
}

// ------------------------------------------------- fused flash attention
// One block per (b*head, 128-query-row slab). 8 waves x 16 rows each.
// S = q(16x32) @ kn^T via WMMA K=32; online softmax; P(16x32) @ V via WMMA.
// K tile staged with async global->LDS (ASYNCcnt) when available, overlapped
// with the V-tile transpose through VGPRs.
__global__ __launch_bounds__(256) void attn_fused(
    const h16* __restrict__ qn, const h16* __restrict__ kn,
    const h16* __restrict__ vv, h16* __restrict__ oh) {
  __shared__ h16 lds_k[32 * 32];       // kn tile, row-major [tok][d]
  __shared__ h16 lds_v[32 * 32];       // v tile transposed [d][tok]
  __shared__ h16 lds_p[8 * 16 * 32];   // per-wave P staging

  const int lane = threadIdx.x & 31;
  const int wv = threadIdx.x >> 5;
  const int bh = blockIdx.x;
  const int b = bh >> 3, head = bh & 7;
  const int row0 = blockIdx.y * 128 + wv * 16;
  const int hs = lane >> 4, nl = lane & 15;

  const h16* qbase = qn + (size_t)bh * NTOK * DH_;
  v16h qf = load_frag(qbase + (size_t)row0 * DH_, DH_);

  v8f o0 = (v8f){}, o1 = (v8f){};
  float mrow[8], lrow[8];
#pragma unroll
  for (int r = 0; r < 8; ++r) { mrow[r] = -1e30f; lrow[r] = 0.f; }

  for (int j0 = 0; j0 < NTOK; j0 += 32) {
    __syncthreads();
    {
      const h16* gk = kn + ((size_t)bh * NTOK + j0) * DH_;
#ifdef HAVE_ASYNC_LDS
      // 2048B contiguous: 128 lanes x b128, waves 0-3 (uniform per wave).
      if (threadIdx.x < 128) {
        __builtin_amdgcn_global_load_async_to_lds_b128(
            (AS1 i32x4*)(void*)(gk + (size_t)threadIdx.x * 8),
            (AS3 i32x4*)(void*)(lds_k + (size_t)threadIdx.x * 8),
            0, 0);
      }
#else
      reinterpret_cast<uint2*>(lds_k)[threadIdx.x] =
          reinterpret_cast<const uint2*>(gk)[threadIdx.x];
#endif
      // v tile: transpose [tok][d] -> [d][tok] through VGPRs (overlaps async)
      union { uint2 u; h16 h[4]; } tv;
      tv.u = reinterpret_cast<const uint2*>(vv + ((size_t)bh * NTOK + j0) * DH_)[threadIdx.x];
      int tok = threadIdx.x >> 3;
      int d0 = (threadIdx.x & 7) * 4;
#pragma unroll
      for (int j = 0; j < 4; ++j) lds_v[(d0 + j) * 32 + tok] = tv.h[j];
#ifdef HAVE_ASYNC_LDS
      __builtin_amdgcn_s_wait_asynccnt(0);
#endif
    }
    __syncthreads();

    v8f s0 = (v8f){}, s1 = (v8f){};
    {
      v16h bk0 = load_frag(lds_k, 32);
      v16h bk1 = load_frag(lds_k + 16 * 32, 32);
      s0 = WMMA_F16(qf, bk0, s0);
      s1 = WMMA_F16(qf, bk1, s1);
    }

    h16* prow = lds_p + wv * (16 * 32);
#pragma unroll
    for (int r = 0; r < 8; ++r) {
      float a0 = s0[r], a1 = s1[r];
      float tm = redmax16(fmaxf(a0, a1));
      float mn = fmaxf(mrow[r], tm);
      float corr = __expf(mrow[r] - mn);
      float p0 = __expf(a0 - mn);
      float p1 = __expf(a1 - mn);
      float ts = redsum16(p0 + p1);
      lrow[r] = lrow[r] * corr + ts;
      mrow[r] = mn;
      o0[r] = o0[r] * corr;
      o1[r] = o1[r] * corr;
      prow[(r + 8 * hs) * 32 + nl] = (h16)p0;
      prow[(r + 8 * hs) * 32 + 16 + nl] = (h16)p1;
    }
    v16h pf = load_frag(prow, 32);
    v16h bv0 = load_frag(lds_v, 32);
    v16h bv1 = load_frag(lds_v + 16 * 32, 32);
    o0 = WMMA_F16(pf, bv0, o0);
    o1 = WMMA_F16(pf, bv1, o1);
  }

#pragma unroll
  for (int r = 0; r < 8; ++r) {
    float inv = 1.f / lrow[r];
    int row = row0 + r + 8 * hs;
    size_t base = ((size_t)(b * NTOK + row)) * C_ + head * DH_;
    oh[base + nl] = (h16)(o0[r] * inv);
    oh[base + 16 + nl] = (h16)(o1[r] * inv);
  }
}

// ------------------------------------------------- elementwise kernels
__global__ __launch_bounds__(256) void cvt_f16(const float* __restrict__ s,
                                               h16* __restrict__ d, int n) {
  int i = blockIdx.x * 256 + threadIdx.x;
  if (i < n) d[i] = (h16)s[i];
}

// depthwise 3x3 + bias + BN1; writes f16 rows in patch-token order
__global__ __launch_bounds__(256) void dwconv_bn(
    const float* __restrict__ x, const float* __restrict__ dww,
    const float* __restrict__ dwb, const float* __restrict__ g,
    const float* __restrict__ bt, const float* __restrict__ mu,
    const float* __restrict__ var, h16* __restrict__ y1h) {
  int idx = blockIdx.x * 256 + threadIdx.x;
  int w = idx & 31, h = (idx >> 5) & 31, c = (idx >> 10) & 255, b = idx >> 18;
  const float* xp = x + ((size_t)(b * C_ + c) << 10);
  float sum = dwb[c];
#pragma unroll
  for (int kh = 0; kh < 3; ++kh) {
    int hh = h + kh - 1;
    if (hh < 0 || hh >= 32) continue;
#pragma unroll
    for (int kw = 0; kw < 3; ++kw) {
      int ww = w + kw - 1;
      if (ww < 0 || ww >= 32) continue;
      sum += xp[hh * 32 + ww] * dww[c * 9 + kh * 3 + kw];
    }
  }
  float v = (sum - mu[c]) * rsqrtf(var[c] + 1e-5f) * g[c] + bt[c];
  int n = (((h >> 1) * 16 + (w >> 1)) * 2 + (h & 1)) * 2 + (w & 1);
  y1h[((size_t)(b * NTOK + n)) * C_ + c] = (h16)v;
}

// LayerNorm over C=256; one wave per row; f16 output
__global__ __launch_bounds__(256) void ln_rows(const float* __restrict__ t,
                                               const float* __restrict__ g,
                                               const float* __restrict__ bta,
                                               h16* __restrict__ hout) {
  int lane = threadIdx.x & 31, wv = threadIdx.x >> 5;
  int m = blockIdx.x * 8 + wv;
  const float* row = t + (size_t)m * C_;
  float v[8];
  float s = 0.f;
#pragma unroll
  for (int j = 0; j < 8; ++j) { v[j] = row[j * 32 + lane]; s += v[j]; }
  float mean = redsum32(s) * (1.f / 256.f);
  float q = 0.f;
#pragma unroll
  for (int j = 0; j < 8; ++j) { float d = v[j] - mean; q += d * d; }
  float rstd = rsqrtf(redsum32(q) * (1.f / 256.f) + 1e-5f);
  h16* orow = hout + (size_t)m * C_;
#pragma unroll
  for (int j = 0; j < 8; ++j) {
    int c = j * 32 + lane;
    orow[c] = (h16)((v[j] - mean) * rstd * g[c] + bta[c]);
  }
}

// l2-normalize q,k per (b,n,head), scale folded into q; v -> f16. One wave each.
__global__ __launch_bounds__(256) void qk_norm(
    const float* __restrict__ q, const float* __restrict__ kv,
    const float* __restrict__ scale, h16* __restrict__ qn,
    h16* __restrict__ kn, h16* __restrict__ vo) {
  int lane = threadIdx.x & 31, wv = threadIdx.x >> 5;
  int idx = blockIdx.x * 8 + wv;          // (b*NTOK+n)*8 + head
  int b = idx >> 13;
  int n = (idx >> 3) & 1023;
  int head = idx & 7;
  size_t rq = ((size_t)(b * NTOK + n)) * C_ + head * DH_ + lane;
  size_t rkv = ((size_t)(b * NTOK + n)) * (2 * C_) + head * DH_ + lane;
  float qv = q[rq];
  float kk = kv[rkv];
  float vvv = kv[rkv + C_];
  float nq = sqrtf(redsum32(qv * qv));
  float nk = sqrtf(redsum32(kk * kk));
  size_t o = (((size_t)(b * HEADS_ + head)) * NTOK + n) * DH_ + lane;
  qn[o] = (h16)(qv / fmaxf(nq, 1e-12f) * scale[head]);
  kn[o] = (h16)(kk / fmaxf(nk, 1e-12f));
  vo[o] = (h16)vvv;
}

// u * gelu(g) from z[m][1536] -> ug[m][768] f16
__global__ __launch_bounds__(256) void geglu(const h16* __restrict__ z,
                                             h16* __restrict__ ug) {
  int idx = blockIdx.x * 256 + threadIdx.x;
  int m = idx / HID_, j = idx % HID_;
  float u = (float)z[(size_t)m * (2 * HID_) + j];
  float g = (float)z[(size_t)m * (2 * HID_) + HID_ + j];
  ug[idx] = (h16)(u * gelu_f(g));
}

// concat[x, feat] in (b,hw) x 512 layout; f32 + f16 copies
__global__ __launch_bounds__(256) void concat_build(
    const float* __restrict__ x, const float* __restrict__ t,
    float* __restrict__ cf, h16* __restrict__ ch) {
  int idx = blockIdx.x * 256 + threadIdx.x;
  int col = idx & 511, m = idx >> 9;
  int b = m >> 10, hw = m & 1023;
  float v;
  if (col < C_) {
    v = x[((size_t)(b * C_ + col) << 10) + hw];
  } else {
    int c = col - C_;
    int h = hw >> 5, w = hw & 31;
    int n = (((h >> 1) * 16 + (w >> 1)) * 2 + (h & 1)) * 2 + (w & 1);
    v = t[((size_t)(b * NTOK + n)) * C_ + c];
  }
  cf[idx] = v;
  ch[idx] = (h16)v;
}

// ---------------------------------------------------------------- host
extern "C" void kernel_launch(void* const* d_in, const int* in_sizes, int n_in,
                              void* d_out, int out_size, void* d_ws,
                              size_t ws_size, hipStream_t stream) {
  const float* x    = (const float*)d_in[0];
  const float* dw_w = (const float*)d_in[1];
  const float* dw_b = (const float*)d_in[2];
  const float* bn1g = (const float*)d_in[3];
  const float* bn1b = (const float*)d_in[4];
  const float* bn1m = (const float*)d_in[5];
  const float* bn1v = (const float*)d_in[6];
  const float* bn2g = (const float*)d_in[7];
  const float* bn2b = (const float*)d_in[8];
  const float* bn2m = (const float*)d_in[9];
  const float* bn2v = (const float*)d_in[10];
  const float* pw_w = (const float*)d_in[11];
  const float* pw_b = (const float*)d_in[12];
  const float* ln1g = (const float*)d_in[13];
  const float* ln1b = (const float*)d_in[14];
  const float* q_w  = (const float*)d_in[15];
  const float* q_b  = (const float*)d_in[16];
  const float* kv_w = (const float*)d_in[17];
  const float* kv_b = (const float*)d_in[18];
  const float* scal = (const float*)d_in[19];
  const float* pj_w = (const float*)d_in[20];
  const float* pj_b = (const float*)d_in[21];
  const float* ln2g = (const float*)d_in[22];
  const float* ln2b = (const float*)d_in[23];
  const float* f1_w = (const float*)d_in[24];
  const float* f1_b = (const float*)d_in[25];
  const float* f2_w = (const float*)d_in[26];
  const float* f2_b = (const float*)d_in[27];
  const float* fa_w = (const float*)d_in[28];
  const float* fa_b = (const float*)d_in[29];
  const float* fco_w= (const float*)d_in[30];
  const float* fco_b= (const float*)d_in[31];

  char* base = (char*)d_ws;
  size_t off = 0;
  auto alloc = [&](size_t bytes) -> void* {
    void* r = base + off;
    off += (bytes + 255) & ~(size_t)255;
    return r;
  };

  h16*   y1h  = (h16*)  alloc((size_t)MROWS * C_ * 2);
  float* tbuf = (float*)alloc((size_t)MROWS * C_ * 4);
  h16*   hbuf = (h16*)  alloc((size_t)MROWS * C_ * 2);
  h16*   ohb  = (h16*)  alloc((size_t)MROWS * C_ * 2);
  // union: {q,kv,qn,kn,v} (attention inputs) overlap with z (fc1 output)
  size_t usave = off;
  float* qbuf = (float*)alloc((size_t)MROWS * C_ * 4);
  float* kvb  = (float*)alloc((size_t)MROWS * 2 * C_ * 4);
  h16*   qnb  = (h16*)  alloc((size_t)MROWS * C_ * 2);
  h16*   knb  = (h16*)  alloc((size_t)MROWS * C_ * 2);
  h16*   vvb  = (h16*)  alloc((size_t)MROWS * C_ * 2);
  size_t uend = off;
  off = usave;
  h16*   zbuf = (h16*)  alloc((size_t)MROWS * 2 * HID_ * 2);
  if (off < uend) off = uend;
  h16*   ugb  = (h16*)  alloc((size_t)MROWS * HID_ * 2);
  float* catf = (float*)alloc((size_t)MROWS * 2 * C_ * 4);
  h16*   cath = (h16*)  alloc((size_t)MROWS * 2 * C_ * 2);
  h16*   cgb  = (h16*)  alloc((size_t)MROWS * 2 * C_ * 2);
  h16*   pwh  = (h16*)  alloc(65536 * 2);
  h16*   qwh  = (h16*)  alloc(131072 * 2);
  h16*   kvwh = (h16*)  alloc(262144 * 2);
  h16*   pjwh = (h16*)  alloc(131072 * 2);
  h16*   f1wh = (h16*)  alloc(786432 * 2);
  h16*   f2wh = (h16*)  alloc(393216 * 2);
  h16*   fawh = (h16*)  alloc(262144 * 2);
  h16*   fcoh = (h16*)  alloc(131072 * 2);

  auto cv = [&](const float* s, h16* d, int n) {
    cvt_f16<<<(n + 255) / 256, 256, 0, stream>>>(s, d, n);
  };
  cv(pw_w, pwh, 65536);   cv(q_w, qwh, 131072);
  cv(kv_w, kvwh, 262144); cv(pj_w, pjwh, 131072);
  cv(f1_w, f1wh, 786432); cv(f2_w, f2wh, 393216);
  cv(fa_w, fawh, 262144); cv(fco_w, fcoh, 131072);

  dwconv_bn<<<(B_ * C_ * HW_) / 256, 256, 0, stream>>>(
      x, dw_w, dw_b, bn1g, bn1b, bn1m, bn1v, y1h);

  dim3 blk(256);
  // pointwise conv + BN2 + GELU -> t (token order)
  gemm_wmma<2><<<dim3(128, 2), blk, 0, stream>>>(
      y1h, pwh, pw_b, tbuf, nullptr, nullptr, nullptr,
      bn2g, bn2b, bn2m, bn2v, MROWS, C_, C_);

  for (int i = 0; i < 2; ++i) {
    ln_rows<<<1024, blk, 0, stream>>>(tbuf, ln1g + i * C_, ln1b + i * C_, hbuf);
    gemm_wmma<0><<<dim3(128, 2), blk, 0, stream>>>(
        hbuf, qwh + i * 65536, q_b + i * C_, qbuf, nullptr, nullptr, nullptr,
        nullptr, nullptr, nullptr, nullptr, MROWS, C_, C_);
    gemm_wmma<0><<<dim3(128, 4), blk, 0, stream>>>(
        hbuf, kvwh + i * 131072, kv_b + i * 2 * C_, kvb, nullptr, nullptr,
        nullptr, nullptr, nullptr, nullptr, nullptr, MROWS, 2 * C_, C_);
    qk_norm<<<8192, blk, 0, stream>>>(qbuf, kvb, scal + i * HEADS_, qnb, knb, vvb);
    attn_fused<<<dim3(B_ * HEADS_, NTOK / 128), blk, 0, stream>>>(qnb, knb, vvb, ohb);
    gemm_wmma<1><<<dim3(128, 2), blk, 0, stream>>>(
        ohb, pjwh + i * 65536, pj_b + i * C_, tbuf, nullptr, tbuf, nullptr,
        nullptr, nullptr, nullptr, nullptr, MROWS, C_, C_);
    ln_rows<<<1024, blk, 0, stream>>>(tbuf, ln2g + i * C_, ln2b + i * C_, hbuf);
    gemm_wmma<3><<<dim3(128, 12), blk, 0, stream>>>(
        hbuf, f1wh + i * 393216, f1_b + i * 2 * HID_, nullptr, zbuf, nullptr,
        nullptr, nullptr, nullptr, nullptr, nullptr, MROWS, 2 * HID_, C_);
    geglu<<<(MROWS * HID_) / 256, blk, 0, stream>>>(zbuf, ugb);
    gemm_wmma<1><<<dim3(128, 2), blk, 0, stream>>>(
        ugb, f2wh + i * 196608, f2_b + i * C_, tbuf, nullptr, tbuf, nullptr,
        nullptr, nullptr, nullptr, nullptr, MROWS, C_, HID_);
  }

  concat_build<<<(MROWS * 2 * C_) / 256, blk, 0, stream>>>(x, tbuf, catf, cath);
  // gate = sigmoid(fa(concat)); cg = gate * concat (f16)
  gemm_wmma<4><<<dim3(128, 4), blk, 0, stream>>>(
      cath, fawh, fa_b, nullptr, cgb, nullptr, catf,
      nullptr, nullptr, nullptr, nullptr, MROWS, 2 * C_, 2 * C_);
  // out = fco(cg) stored NCHW
  gemm_wmma<5><<<dim3(128, 2), blk, 0, stream>>>(
      cgb, fcoh, fco_b, (float*)d_out, nullptr, nullptr, nullptr,
      nullptr, nullptr, nullptr, nullptr, MROWS, C_, 2 * C_);
}
